// GraphConvolution_9758165697084
// MI455X (gfx1250) — matrix operand order repaired and checked by reading, not compiled, not run on hardware
//
#include <hip/hip_runtime.h>
#include <math.h>

// Problem constants (match the reference).
#define BB 2
#define NN 50000
#define DD 64
#define EE 800000
#define HH 32
#define BN (BB * NN)   // 100000 rows

typedef __attribute__((ext_vector_type(16))) _Float16 v16h;
typedef __attribute__((ext_vector_type(8)))  float    v8f;

__device__ __forceinline__ float elu1(float x) {
    return x > 0.f ? x : (__expf(x) - 1.f);
}
__device__ __forceinline__ float leaky(float x, float s) {
    return x >= 0.f ? x : s * x;
}
// Monotonic float<->uint mapping so unsigned atomicMax == float max.
__device__ __forceinline__ unsigned f2key(float f) {
    unsigned u = __float_as_uint(f);
    return (u & 0x80000000u) ? ~u : (u | 0x80000000u);
}
__device__ __forceinline__ float key2f(unsigned k) {
    return __uint_as_float((k & 0x80000000u) ? (k & 0x7FFFFFFFu) : ~k);
}

// ---------------------------------------------------------------------------
// Kernel A: tf_feat = x_influence(100000x64) @ W(64x64), f16 WMMA, f32 accum.
// Block = 128 (4 waves); each wave owns one 16-wide n-tile of a 16-row m-tile.
// K=64 -> two v_wmma_f32_16x16x32_f16 per wave. No tails, EXEC all ones.
// VGPR layouts follow CDNA5 ISA 7.12.2 (wave32).
// ---------------------------------------------------------------------------
__global__ void tf_gemm_wmma(const float* __restrict__ X,
                             const float* __restrict__ W,
                             float* __restrict__ TF) {
    const int lane = threadIdx.x & 31;
    const int wave = threadIdx.x >> 5;          // 0..3 -> n-tile
    const int m0   = blockIdx.x * 16;
    const int n0   = wave * 16;
    const int half = lane >> 4;                 // lane group 0/1
    const int l15  = lane & 15;

    const float* xrow = X + (size_t)(m0 + l15) * DD;   // A row for this lane
    v8f c = {};

#pragma unroll
    for (int kb = 0; kb < 2; ++kb) {
        const int kbase = kb * 32;
        v16h a, b;
#pragma unroll
        for (int i = 0; i < 16; ++i) {
            // A 16x32 f16 layout: lanes 0-15 hold K {0..7,16..23}, lanes 16-31
            // hold K {8..15,24..31} for rows M = lane&15.
            const int ka = kbase + (i & 7) + 8 * half + ((i >= 8) ? 16 : 0);
            a[i] = (_Float16)xrow[ka];
            // B 32x16 f16 layout: N = lane&15; lanes 0-15 K {0..7,16..23}, lanes 16-31
            // K {8..15,24..31}, same K order as A.
            const int kbv = kbase + (i & 7) + 8 * half + ((i >= 8) ? 16 : 0);
            b[i] = (_Float16)W[kbv * DD + n0 + l15];
        }
        c = __builtin_amdgcn_wmma_f32_16x16x32_f16(
                /*neg_a=*/false, a, /*neg_b=*/false, b,
                /*c_mod=*/(short)0, c, /*reuse_a=*/false, /*reuse_b=*/false);
    }

    // C/D layout: VGPR r -> M = r + 8*half, N = lane&15.
#pragma unroll
    for (int r = 0; r < 8; ++r) {
        const int row = m0 + r + 8 * half;
        TF[(size_t)row * DD + n0 + l15] = c[r];
    }
}

// ---------------------------------------------------------------------------
// Init: den = 0, running-max key = encode(-inf).
// ---------------------------------------------------------------------------
__global__ void init_ws(unsigned* __restrict__ mkey, float* __restrict__ den) {
    const int t = blockIdx.x * blockDim.x + threadIdx.x;
    if (t < BN) {
        mkey[t] = 0x007FFFFFu;   // f2key(-inf)
        den[t]  = 0.f;
    }
}

// ---------------------------------------------------------------------------
// Kernel B: per-node pass. One wave per node p in [0,BN).
// 4 attention dots via butterfly reduction; H=32 gate MLP maps to 32 lanes.
// ---------------------------------------------------------------------------
__global__ void node_pass(const float* __restrict__ TF,
                          const float* __restrict__ x_state,
                          const float* __restrict__ sbeta,   // (128)
                          const float* __restrict__ iattn,   // (128)
                          const float* __restrict__ w1, const float* __restrict__ b1,
                          const float* __restrict__ w2, const float* __restrict__ b2,
                          float* __restrict__ s_r, float* __restrict__ s_c,
                          float* __restrict__ e_r, float* __restrict__ e_c,
                          float* __restrict__ filt) {
    const int lane = threadIdx.x & 31;
    const int p    = blockIdx.x * 8 + (threadIdx.x >> 5);
    if (p >= BN) return;

    const float* tf = TF + (size_t)p * DD;
    const float t0 = tf[lane], t1 = tf[lane + 32];

    float sr = t0 * sbeta[lane]      + t1 * sbeta[lane + 32];
    float sc = t0 * sbeta[64 + lane] + t1 * sbeta[96 + lane];
    float er = t0 * iattn[lane]      + t1 * iattn[lane + 32];
    float ec = t0 * iattn[64 + lane] + t1 * iattn[96 + lane];
#pragma unroll
    for (int m = 16; m; m >>= 1) {
        sr += __shfl_xor(sr, m, 32);
        sc += __shfl_xor(sc, m, 32);
        er += __shfl_xor(er, m, 32);
        ec += __shfl_xor(ec, m, 32);
    }
    if (lane == 0) { s_r[p] = sr; s_c[p] = sc; e_r[p] = er; e_c[p] = ec; }

    // gate = elu(elu(x*w1 + b1) @ w2 + b2); lane = hidden unit
    const float xs = x_state[p];
    float h = elu1(xs * w1[lane] + b1[lane]) * w2[lane];
#pragma unroll
    for (int m = 16; m; m >>= 1) h += __shfl_xor(h, m, 32);
    const float gate = elu1(h + b2[0]);

    float* fo = filt + (size_t)p * DD;
    fo[lane]      = t0 * gate;
    fo[lane + 32] = t1 * gate;
}

// ---------------------------------------------------------------------------
// Kernel C: edge pass 1 — state message scatter-add + segment float-max.
// ---------------------------------------------------------------------------
__global__ void edge_pass1(const int* __restrict__ Lidx,
                           const float* __restrict__ s_r, const float* __restrict__ s_c,
                           const float* __restrict__ e_r, const float* __restrict__ e_c,
                           const float* __restrict__ x_state,
                           float* __restrict__ s_accum,      // d_out state region
                           unsigned* __restrict__ mkey) {
    const int e = blockIdx.x * blockDim.x + threadIdx.x;
    if (e >= EE) return;
    const int row = Lidx[2 * e], col = Lidx[2 * e + 1];
#pragma unroll
    for (int b = 0; b < BB; ++b) {
        const int pr = b * NN + row, pc = b * NN + col;
        const float ls = leaky(s_r[pr] + s_c[pc], 0.02f);
        atomicAdd(&s_accum[pr], ls * x_state[pc]);
        const float le = leaky(e_r[pr] + e_c[pc], 0.2f);
        atomicMax(&mkey[pr], f2key(le));
    }
}

// ---------------------------------------------------------------------------
// Kernel D: edge pass 2 — unnormalized softmax-weighted scatter.
// One wave per edge; lanes cover D=64 in two strides. Coalesced 256B
// gathers of filt[col,:] and 256B GLOBAL_ATOMIC_ADD_F32 into E_accum[row,:].
// ---------------------------------------------------------------------------
__global__ void edge_pass2(const int* __restrict__ Lidx,
                           const float* __restrict__ e_r, const float* __restrict__ e_c,
                           const unsigned* __restrict__ mkey,
                           const float* __restrict__ filt,
                           float* __restrict__ den,
                           float* __restrict__ e_accum) {   // d_out influence region
    const int lane = threadIdx.x & 31;
    const int e = blockIdx.x * (blockDim.x >> 5) + (threadIdx.x >> 5);
    if (e >= EE) return;
    const int row = Lidx[2 * e], col = Lidx[2 * e + 1];
#pragma unroll
    for (int b = 0; b < BB; ++b) {
        const int pr = b * NN + row, pc = b * NN + col;
        const float le = leaky(e_r[pr] + e_c[pc], 0.2f);
        const float ex = __expf(le - key2f(mkey[pr]));
        if (lane == 0) atomicAdd(&den[pr], ex);
        const float* fc = filt + (size_t)pc * DD;
        float* ea = e_accum + (size_t)pr * DD;
        atomicAdd(&ea[lane],      ex * fc[lane]);
        atomicAdd(&ea[lane + 32], ex * fc[lane + 32]);
    }
}

// ---------------------------------------------------------------------------
// Kernel E: finalize (in place over d_out accumulators).
// ---------------------------------------------------------------------------
__global__ void finalize(const float* __restrict__ TF,
                         const float* __restrict__ den,
                         const float* __restrict__ x_state,
                         const float* __restrict__ Xs,
                         const float* __restrict__ self_act,
                         const float* __restrict__ sws, const float* __restrict__ swn,
                         const float* __restrict__ iws, const float* __restrict__ iwn,
                         float* __restrict__ out_state,
                         float* __restrict__ out_infl) {
    const size_t t = (size_t)blockIdx.x * blockDim.x + threadIdx.x;
    if (t >= (size_t)BN * DD) return;
    const int p = (int)(t >> 6);
    const int d = (int)(t & 63);

    const float dn  = den[p];
    const float acc = out_infl[t];
    const float en  = dn > 0.f ? acc / dn : 0.f;   // softmax normalize; empty seg -> 0
    out_infl[t] = elu1(iws[0] * TF[t] + iwn[0] * en);

    if (d == 0) {
        const int n = p % NN;
        const float sn  = out_state[p] + self_act[n];
        const float su  = elu1(sws[0] * x_state[p] + swn[0] * sn);
        const float xsv = Xs[p];
        out_state[p] = su * (1.f - xsv) + xsv;
    }
}

// ---------------------------------------------------------------------------
extern "C" void kernel_launch(void* const* d_in, const int* in_sizes, int n_in,
                              void* d_out, int out_size, void* d_ws, size_t ws_size,
                              hipStream_t stream) {
    (void)in_sizes; (void)n_in; (void)ws_size;

    const float* x_state = (const float*)d_in[0];
    const float* x_infl  = (const float*)d_in[1];
    const int*   Lidx    = (const int*)  d_in[2];
    // d_in[3] = L_values: unused by the reference.
    const float* self_a  = (const float*)d_in[4];
    const float* Xs      = (const float*)d_in[5];
    const float* W       = (const float*)d_in[6];
    const float* sbeta   = (const float*)d_in[7];
    const float* sws     = (const float*)d_in[8];
    const float* swn     = (const float*)d_in[9];
    const float* w1      = (const float*)d_in[10];
    const float* b1      = (const float*)d_in[11];
    const float* w2      = (const float*)d_in[12];
    const float* b2      = (const float*)d_in[13];
    const float* iattn   = (const float*)d_in[14];
    const float* iws     = (const float*)d_in[15];
    const float* iwn     = (const float*)d_in[16];

    float* out_state = (float*)d_out;        // BN floats (doubles as S_neigh accum)
    float* out_infl  = out_state + BN;       // BN*DD floats (doubles as E_neigh accum)

    // Workspace layout (~54 MB).
    float* tf    = (float*)d_ws;                      // BN*DD
    float* filt  = tf   + (size_t)BN * DD;            // BN*DD
    float* s_r   = filt + (size_t)BN * DD;            // BN
    float* s_c   = s_r + BN;
    float* e_r   = s_c + BN;
    float* e_c   = e_r + BN;
    float* den   = e_c + BN;
    unsigned* mkey = (unsigned*)(den + BN);

    hipMemsetAsync(d_out, 0, (size_t)out_size * sizeof(float), stream);
    init_ws<<<(BN + 255) / 256, 256, 0, stream>>>(mkey, den);

    tf_gemm_wmma<<<BN / 16, 128, 0, stream>>>(x_infl, W, tf);
    node_pass<<<BN / 8, 256, 0, stream>>>(tf, x_state, sbeta, iattn,
                                          w1, b1, w2, b2,
                                          s_r, s_c, e_r, e_c, filt);
    edge_pass1<<<(EE + 255) / 256, 256, 0, stream>>>(Lidx, s_r, s_c, e_r, e_c,
                                                     x_state, out_state, mkey);
    edge_pass2<<<EE / 8, 256, 0, stream>>>(Lidx, e_r, e_c, mkey, filt, den, out_infl);
    finalize<<<((size_t)BN * DD) / 256, 256, 0, stream>>>(tf, den, x_state, Xs, self_a,
                                                          sws, swn, iws, iwn,
                                                          out_state, out_infl);
}
